// point_encoder_66958540145286
// MI455X (gfx1250) — matrix operand order repaired and checked by reading, not compiled, not run on hardware
//
#include <hip/hip_runtime.h>
#include <hip/hip_bf16.h>

// ---------------- problem constants ----------------
#define Nn 50000
#define Ee 400000
#define Dd 128
#define Hh 4
#define DKk 32
#define NRr 50
#define NRPAD 64
#define Ll 2
#define NGg 100
#define TEe 32
#define OUTo 512

typedef __attribute__((ext_vector_type(16))) __bf16 v16bf;
typedef __attribute__((ext_vector_type(8)))  float  v8f;

__device__ __forceinline__ float geluf(float x) {
    float x3 = x * x * x;
    return 0.5f * x * (1.0f + tanhf(0.7978845608028654f * (x + 0.044715f * x3)));
}

__device__ __forceinline__ void atomicMaxFloat(float* addr, float value) {
    if (value >= 0.0f)
        atomicMax((int*)addr, __float_as_int(value));
    else
        atomicMin((unsigned int*)addr, __float_as_uint(value));
}

// ---------------- generic bf16 WMMA GEMM ----------------
// C[M,Ncols] = A[M,K] (bf16, row-major) x Bt[Ncols,K] (bf16, row-major = B^T)
// Block: 256 threads = 8 waves; block tile 16(M) x 128(N); wave tile 16x16.
// EPI: 0 store f32; 1 store bf16; 2 Cf += acc; 3 bf16 = gelu(acc+bias);
//      4 Cf += acc+bias, Cb = bf16(Cf); 5 Cf = acc+bias
template<int EPI>
__global__ __launch_bounds__(256) void gemm_bf16(
        const __bf16* __restrict__ A, const __bf16* __restrict__ Bt,
        int M, int K, int Ncols,
        const float* __restrict__ bias,
        float* __restrict__ Cf, __bf16* __restrict__ Cb)
{
    __shared__ __bf16 As[16 * 128];

    const int tid  = threadIdx.x;
    const int m0   = blockIdx.x * 16;
    const int wv   = tid >> 5;
    const int lane = tid & 31;
    const int n0   = blockIdx.y * 128 + wv * 16;

    // cooperative A-tile load into LDS (zero-pad rows >= M)
    const int halfK = K >> 1;                    // K is even (64 or 128)
    unsigned int* As32 = (unsigned int*)As;
    const unsigned int* A32 = (const unsigned int*)A;
    for (int i = tid; i < 8 * K; i += 256) {
        int row  = i / halfK;
        int kk   = i - row * halfK;
        int grow = m0 + row;
        unsigned int val = 0u;
        if (grow < M) val = A32[(long)grow * halfK + kk];
        As32[i] = val;
    }
    __syncthreads();

    v8f acc = {0.f, 0.f, 0.f, 0.f, 0.f, 0.f, 0.f, 0.f};

    const int arow = lane & 15;
    const int akb  = (lane >> 4) << 3;   // 0 / 8   (16-bit A fragment layout)
    const int bkb  = (lane >> 4) << 3;   // 0 / 8   (16-bit B fragment layout)
    const int bcol = n0 + (lane & 15);
    const __bf16* Bp = Bt + (long)bcol * K;

    for (int k0 = 0; k0 < K; k0 += 32) {
        union { v16bf v; uint4 u[2]; } fa, fb;
        const __bf16* pa = As + arow * K + k0 + akb;
        fa.u[0] = *(const uint4*)(pa);
        fa.u[1] = *(const uint4*)(pa + 16);
        const __bf16* pb = Bp + k0 + bkb;
        __builtin_prefetch(pb + 64, 0, 0);
        fb.u[0] = *(const uint4*)(pb);
        fb.u[1] = *(const uint4*)(pb + 16);
        acc = __builtin_amdgcn_wmma_f32_16x16x32_bf16(
                  false, fa.v, false, fb.v, (short)0, acc, false, false);
    }

    // epilogue: lanes 0-15 rows m0+0..7, lanes 16-31 rows m0+8..15
    const int row0 = m0 + ((lane >> 4) << 3);
    const int col  = bcol;
    float bv = 0.0f;
    if (EPI == 3 || EPI == 4 || EPI == 5) bv = bias[col];
#pragma unroll
    for (int r = 0; r < 8; ++r) {
        int row = row0 + r;
        if (row >= M) continue;
        long idx = (long)row * Ncols + col;
        float v = acc[r];
        if (EPI == 0) {
            Cf[idx] = v;
        } else if (EPI == 1) {
            Cb[idx] = (__bf16)v;
        } else if (EPI == 2) {
            Cf[idx] += v;
        } else if (EPI == 3) {
            Cb[idx] = (__bf16)geluf(v + bv);
        } else if (EPI == 4) {
            float t = Cf[idx] + v + bv;
            Cf[idx] = t;
            Cb[idx] = (__bf16)t;
        } else if (EPI == 5) {
            Cf[idx] = v + bv;
        }
    }
}

// ---------------- helper kernels ----------------
__global__ void k_fill_f32(float* p, float v, long n) {
    long i = (long)blockIdx.x * blockDim.x + threadIdx.x;
    for (; i < n; i += (long)gridDim.x * blockDim.x) p[i] = v;
}

__global__ void k_cvt_bf16(const float* __restrict__ s, __bf16* __restrict__ d, long n) {
    long i = (long)blockIdx.x * blockDim.x + threadIdx.x;
    for (; i < n; i += (long)gridDim.x * blockDim.x) d[i] = (__bf16)s[i];
}

// dst[c*Rpad + r] = bf16(src[r*C + c]), zero pad r >= R
__global__ void k_transpose_pad(const float* __restrict__ src, __bf16* __restrict__ dst,
                                int R, int C, int Rpad) {
    long t = (long)blockIdx.x * blockDim.x + threadIdx.x;
    long total = (long)C * Rpad;
    if (t >= total) return;
    int c = (int)(t / Rpad);
    int r = (int)(t - (long)c * Rpad);
    dst[t] = (__bf16)((r < R) ? src[(long)r * C + c] : 0.0f);
}

// x[n,:] = concat(emb[atom_types[n]], pos[n]) @ W_init + b_init
__global__ __launch_bounds__(128) void k_init_nodes(
        const float* __restrict__ pos, const int* __restrict__ at,
        const float* __restrict__ emb, const float* __restrict__ Wi,
        const float* __restrict__ bi, float* __restrict__ x, __bf16* __restrict__ xb)
{
    __shared__ float s0[TEe + 3];
    int n = blockIdx.x;
    int t = threadIdx.x;
    if (t < TEe)               s0[t] = emb[(long)at[n] * TEe + t];
    else if (t < TEe + 3)      s0[t] = pos[(long)n * 3 + (t - TEe)];
    __syncthreads();
    float acc = bi[t];
#pragma unroll
    for (int i = 0; i < TEe + 3; ++i) acc += s0[i] * Wi[i * Dd + t];
    x [(long)n * Dd + t] = acc;
    xb[(long)n * Dd + t] = (__bf16)acc;
}

// radial basis per edge, stored bf16 [E, 64] (K padded 50->64)
__global__ void k_rbf(const float* __restrict__ pos, const int* __restrict__ ei,
                      __bf16* __restrict__ rb) {
    long t = (long)blockIdx.x * blockDim.x + threadIdx.x;
    if (t >= (long)Ee * NRPAD) return;
    int e = (int)(t >> 6);
    int r = (int)(t & 63);
    int src = ei[e], dst = ei[Ee + e];
    float dx = pos[src * 3 + 0] - pos[dst * 3 + 0] + 1e-8f;
    float dy = pos[src * 3 + 1] - pos[dst * 3 + 1] + 1e-8f;
    float dz = pos[src * 3 + 2] - pos[dst * 3 + 2] + 1e-8f;
    float d = sqrtf(dx * dx + dy * dy + dz * dz);
    float val = 0.0f;
    if (r < NRr) {
        const float CUT = 6.0f;
        float mu = CUT * (float)r / (float)(NRr - 1);
        float gamma = (NRr / CUT) * (NRr / CUT);
        float dc = fminf(d, CUT);
        float env = 0.5f * (cosf(3.14159265358979323846f * dc / CUT) + 1.0f);
        float dd = d - mu;
        val = expf(-gamma * dd * dd) * env;
    }
    rb[t] = (__bf16)val;
}

// per-(edge,head) attention logit + segment max
__global__ void k_logits(const int* __restrict__ ei, const float* __restrict__ qn,
                         const float* __restrict__ kn, const __bf16* __restrict__ ef,
                         float* __restrict__ abuf, float* __restrict__ smax) {
    long t = (long)blockIdx.x * blockDim.x + threadIdx.x;
    if (t >= (long)Ee * Hh) return;
    int e = (int)(t >> 2);
    int h = (int)(t & 3);
    int src = ei[e], dst = ei[Ee + e];
    const float*  q  = qn + (long)dst * Dd + h * DKk;
    const float*  k  = kn + (long)src * Dd + h * DKk;
    const __bf16* ee = ef + (long)e  * Dd + h * DKk;
    float s = 0.0f;
#pragma unroll
    for (int i = 0; i < DKk; ++i) s += q[i] * (k[i] + (float)ee[i]);
    s *= 0.17677669529663687f;   // 1/sqrt(32)
    abuf[t] = s;
    atomicMaxFloat(&smax[(long)dst * Hh + h], s);
}

// a = exp(logit - max); segment sum
__global__ void k_expsum(const int* __restrict__ ei, float* __restrict__ abuf,
                         const float* __restrict__ smax, float* __restrict__ ssum) {
    long t = (long)blockIdx.x * blockDim.x + threadIdx.x;
    if (t >= (long)Ee * Hh) return;
    int e = (int)(t >> 2);
    int h = (int)(t & 3);
    int dst = ei[Ee + e];
    float a = expf(abuf[t] - smax[(long)dst * Hh + h]);
    abuf[t] = a;
    atomicAdd(&ssum[(long)dst * Hh + h], a);
}

// scatter-add normalized messages: msg[dst,j] += a/(s+eps) * (v[src,j] + e[edge,j])
__global__ void k_message(const int* __restrict__ ei, const float* __restrict__ abuf,
                          const float* __restrict__ ssum, const float* __restrict__ vn,
                          const __bf16* __restrict__ ef, float* __restrict__ msg) {
    long t = (long)blockIdx.x * blockDim.x + threadIdx.x;
    if (t >= (long)Ee * Dd) return;
    int e = (int)(t >> 7);
    int j = (int)(t & 127);
    int h = j >> 5;
    int src = ei[e], dst = ei[Ee + e];
    float w = abuf[(long)e * Hh + h] / (ssum[(long)dst * Hh + h] + 1e-9f);
    float val = w * (vn[(long)src * Dd + j] + (float)ef[(long)e * Dd + j]);
    atomicAdd(&msg[(long)dst * Dd + j], val);
}

// wave-per-row layernorm -> bf16
__global__ __launch_bounds__(256) void k_layernorm(
        const float* __restrict__ x, const float* __restrict__ g,
        const float* __restrict__ b, __bf16* __restrict__ h) {
    int node = blockIdx.x * 8 + (threadIdx.x >> 5);
    int lane = threadIdx.x & 31;
    if (node >= Nn) return;
    float4 xv = *(const float4*)(x + (long)node * Dd + lane * 4);
    float s  = xv.x + xv.y + xv.z + xv.w;
    float s2 = xv.x * xv.x + xv.y * xv.y + xv.z * xv.z + xv.w * xv.w;
    for (int m = 16; m > 0; m >>= 1) {
        s  += __shfl_xor(s,  m, 32);
        s2 += __shfl_xor(s2, m, 32);
    }
    float mean = s * (1.0f / Dd);
    float var  = s2 * (1.0f / Dd) - mean * mean;
    float inv  = rsqrtf(var + 1e-5f);
    int j = lane * 4;
    float xs[4] = {xv.x, xv.y, xv.z, xv.w};
#pragma unroll
    for (int i = 0; i < 4; ++i)
        h[(long)node * Dd + j + i] = (__bf16)((xs[i] - mean) * inv * g[j + i] + b[j + i]);
}

// graph pooling: g[batch[n], j] += x[n, j]
__global__ void k_graph_sum(const int* __restrict__ batch, const float* __restrict__ x,
                            float* __restrict__ g) {
    long t = (long)blockIdx.x * blockDim.x + threadIdx.x;
    if (t >= (long)Nn * Dd) return;
    int n = (int)(t >> 7);
    int j = (int)(t & 127);
    atomicAdd(&g[(long)batch[n] * Dd + j], x[t]);
}

// ---------------- host side ----------------
static inline long alignup(long v) { return (v + 255) & ~255L; }

extern "C" void kernel_launch(void* const* d_in, const int* in_sizes, int n_in,
                              void* d_out, int out_size, void* d_ws, size_t ws_size,
                              hipStream_t stream) {
    const float* pos   = (const float*)d_in[0];
    const int*   atype = (const int*)  d_in[1];
    const int*   eidx  = (const int*)  d_in[2];
    const int*   batch = (const int*)  d_in[3];
    const float* emb   = (const float*)d_in[4];
    const float* W_init= (const float*)d_in[5];
    const float* b_init= (const float*)d_in[6];
    const float* Wq    = (const float*)d_in[7];
    const float* Wk    = (const float*)d_in[8];
    const float* Wv    = (const float*)d_in[9];
    const float* We    = (const float*)d_in[10];
    const float* Wo    = (const float*)d_in[11];
    const float* Wm1   = (const float*)d_in[12];
    const float* bm1   = (const float*)d_in[13];
    const float* Wm2   = (const float*)d_in[14];
    const float* bm2   = (const float*)d_in[15];
    const float* ln_g  = (const float*)d_in[16];
    const float* ln_b  = (const float*)d_in[17];
    const float* W_out = (const float*)d_in[18];
    const float* b_out = (const float*)d_in[19];
    float* out = (float*)d_out;

    // ---- workspace layout ----
    unsigned char* ws = (unsigned char*)d_ws;
    long off = 0;
    auto take = [&](long bytes) { long o = off; off += alignup(bytes); return o; };
    float*  x     = (float*) (ws + take((long)Nn * Dd * 4));
    __bf16* xb    = (__bf16*)(ws + take((long)Nn * Dd * 2));
    __bf16* rb    = (__bf16*)(ws + take((long)Ee * NRPAD * 2));
    float*  qn    = (float*) (ws + take((long)Nn * Dd * 4));
    float*  kn    = (float*) (ws + take((long)Nn * Dd * 4));
    float*  vn    = (float*) (ws + take((long)Nn * Dd * 4));
    __bf16* ef    = (__bf16*)(ws + take((long)Ee * Dd * 2));
    float*  abuf  = (float*) (ws + take((long)Ee * Hh * 4));
    float*  smax  = (float*) (ws + take((long)Nn * Hh * 4));
    float*  ssum  = (float*) (ws + take((long)Nn * Hh * 4));
    float*  msg   = (float*) (ws + take((long)Nn * Dd * 4));
    __bf16* msgb  = (__bf16*)(ws + take((long)Nn * Dd * 2));
    __bf16* hb    = (__bf16*)(ws + take((long)Nn * Dd * 2));
    __bf16* h2b   = (__bf16*)(ws + take((long)Nn * Dd * 2));
    float*  gsum  = (float*) (ws + take((long)NGg * Dd * 4));
    __bf16* gb    = (__bf16*)(ws + take((long)NGg * Dd * 2));
    __bf16* wq_t  = (__bf16*)(ws + take((long)Ll * Dd * Dd * 2));
    __bf16* wk_t  = (__bf16*)(ws + take((long)Ll * Dd * Dd * 2));
    __bf16* wv_t  = (__bf16*)(ws + take((long)Ll * Dd * Dd * 2));
    __bf16* wo_t  = (__bf16*)(ws + take((long)Ll * Dd * Dd * 2));
    __bf16* wm1_t = (__bf16*)(ws + take((long)Ll * Dd * Dd * 2));
    __bf16* wm2_t = (__bf16*)(ws + take((long)Ll * Dd * Dd * 2));
    __bf16* we_t  = (__bf16*)(ws + take((long)Ll * Dd * NRPAD * 2));
    __bf16* wout_t= (__bf16*)(ws + take((long)OUTo * Dd * 2));
    (void)ws_size; (void)in_sizes; (void)n_in; (void)out_size;

    // ---- weight bf16 transposes ----
    {
        int tp = 256;
        long nDD = (long)Dd * Dd;
        dim3 gDD((unsigned)((nDD + tp - 1) / tp));
        for (int l = 0; l < Ll; ++l) {
            k_transpose_pad<<<gDD, tp, 0, stream>>>(Wq  + l * Dd * Dd, wq_t  + l * Dd * Dd, Dd, Dd, Dd);
            k_transpose_pad<<<gDD, tp, 0, stream>>>(Wk  + l * Dd * Dd, wk_t  + l * Dd * Dd, Dd, Dd, Dd);
            k_transpose_pad<<<gDD, tp, 0, stream>>>(Wv  + l * Dd * Dd, wv_t  + l * Dd * Dd, Dd, Dd, Dd);
            k_transpose_pad<<<gDD, tp, 0, stream>>>(Wo  + l * Dd * Dd, wo_t  + l * Dd * Dd, Dd, Dd, Dd);
            k_transpose_pad<<<gDD, tp, 0, stream>>>(Wm1 + l * Dd * Dd, wm1_t + l * Dd * Dd, Dd, Dd, Dd);
            k_transpose_pad<<<gDD, tp, 0, stream>>>(Wm2 + l * Dd * Dd, wm2_t + l * Dd * Dd, Dd, Dd, Dd);
            long nWE = (long)Dd * NRPAD;
            k_transpose_pad<<<dim3((unsigned)((nWE + tp - 1) / tp)), tp, 0, stream>>>(
                We + l * NRr * Dd, we_t + l * Dd * NRPAD, NRr, Dd, NRPAD);
        }
        long nWO = (long)OUTo * Dd;
        k_transpose_pad<<<dim3((unsigned)((nWO + tp - 1) / tp)), tp, 0, stream>>>(
            W_out, wout_t, Dd, OUTo, Dd);
    }

    // ---- node init + edge RBF ----
    k_init_nodes<<<Nn, 128, 0, stream>>>(pos, atype, emb, W_init, b_init, x, xb);
    {
        long n = (long)Ee * NRPAD;
        k_rbf<<<dim3((unsigned)((n + 255) / 256)), 256, 0, stream>>>(pos, eidx, rb);
    }

    const dim3 gemmN((Nn + 15) / 16, 1);   // [N,128] x [128,128]
    const dim3 gemmE((Ee + 15) / 16, 1);   // [E,64]  x [64,128]

    for (int l = 0; l < Ll; ++l) {
        // q/k/v projections (WMMA)
        gemm_bf16<0><<<gemmN, 256, 0, stream>>>(xb, wq_t + l * Dd * Dd, Nn, Dd, Dd, nullptr, qn, nullptr);
        gemm_bf16<0><<<gemmN, 256, 0, stream>>>(xb, wk_t + l * Dd * Dd, Nn, Dd, Dd, nullptr, kn, nullptr);
        gemm_bf16<0><<<gemmN, 256, 0, stream>>>(xb, wv_t + l * Dd * Dd, Nn, Dd, Dd, nullptr, vn, nullptr);
        // edge features e = rb @ We (WMMA), bf16 out
        gemm_bf16<1><<<gemmE, 256, 0, stream>>>(rb, we_t + l * Dd * NRPAD, Ee, NRPAD, Dd, nullptr, nullptr, ef);

        // segmented softmax over incoming edges
        long nh = (long)Nn * Hh;
        k_fill_f32<<<dim3(512), 256, 0, stream>>>(smax, -3.0e38f, nh);
        k_fill_f32<<<dim3(512), 256, 0, stream>>>(ssum, 0.0f, nh);
        long eh = (long)Ee * Hh;
        k_logits<<<dim3((unsigned)((eh + 255) / 256)), 256, 0, stream>>>(eidx, qn, kn, ef, abuf, smax);
        k_expsum<<<dim3((unsigned)((eh + 255) / 256)), 256, 0, stream>>>(eidx, abuf, smax, ssum);

        // message aggregation
        long nd = (long)Nn * Dd;
        k_fill_f32<<<dim3(2048), 256, 0, stream>>>(msg, 0.0f, nd);
        long ed = (long)Ee * Dd;
        k_message<<<dim3((unsigned)((ed + 255) / 256)), 256, 0, stream>>>(eidx, abuf, ssum, vn, ef, msg);
        k_cvt_bf16<<<dim3(2048), 256, 0, stream>>>(msg, msgb, nd);

        // x += msg @ Wo  (WMMA residual)
        gemm_bf16<2><<<gemmN, 256, 0, stream>>>(msgb, wo_t + l * Dd * Dd, Nn, Dd, Dd, nullptr, x, nullptr);

        // pre-norm FC block
        k_layernorm<<<Nn / 8, 256, 0, stream>>>(x, ln_g + l * Dd, ln_b + l * Dd, hb);
        gemm_bf16<3><<<gemmN, 256, 0, stream>>>(hb,  wm1_t + l * Dd * Dd, Nn, Dd, Dd, bm1 + l * Dd, nullptr, h2b);
        gemm_bf16<4><<<gemmN, 256, 0, stream>>>(h2b, wm2_t + l * Dd * Dd, Nn, Dd, Dd, bm2 + l * Dd, x, xb);
    }

    // graph pooling + output projection (WMMA)
    k_fill_f32<<<dim3(64), 256, 0, stream>>>(gsum, 0.0f, (long)NGg * Dd);
    {
        long nd = (long)Nn * Dd;
        k_graph_sum<<<dim3((unsigned)((nd + 255) / 256)), 256, 0, stream>>>(batch, x, gsum);
    }
    k_cvt_bf16<<<dim3(64), 256, 0, stream>>>(gsum, gb, (long)NGg * Dd);
    gemm_bf16<5><<<dim3((NGg + 15) / 16, OUTo / 128), 256, 0, stream>>>(
        gb, wout_t, NGg, Dd, OUTo, b_out, out, nullptr);
}
